// Attention_Fusion_512_39092792328565
// MI455X (gfx1250) — matrix operand order repaired and checked by reading, not compiled, not run on hardware
//
#include <hip/hip_runtime.h>
#include <hip/hip_bf16.h>
#include <stdint.h>

typedef unsigned short u16;
typedef unsigned int   u32;
typedef __attribute__((ext_vector_type(16))) __bf16 v16bf;
typedef __attribute__((ext_vector_type(8)))  float  v8f;
typedef __attribute__((ext_vector_type(4)))  u32    u32x4;
typedef __attribute__((ext_vector_type(8)))  int    i32x8;
typedef __attribute__((ext_vector_type(4)))  int    i32x4;

// Problem constants
#define BB    8
#define CCH   768
#define DMOD  512
#define HWSZ  1024
#define LQ    2048
#define NHEAD 8
#define DHEAD 64

// TDM availability (device pass only; host pass sees 0 and strips the calls)
#if defined(__has_builtin)
#if __has_builtin(__builtin_amdgcn_tensor_load_to_lds)
#define HAVE_TDM 1
#endif
#endif
#ifndef HAVE_TDM
#define HAVE_TDM 0
#endif

#if HAVE_TDM
#define ASYNC_PER_ITER 4  // A tile via 4 async b128/thread; B tile via TDM
#else
#define ASYNC_PER_ITER 6  // A (4) + B (2) tiles via async
#endif

__device__ __forceinline__ u16 f2bf(float f) {
  unsigned u = __builtin_bit_cast(unsigned, f);
  unsigned r = (u + 0x7FFFu + ((u >> 16) & 1u)) >> 16;
  return (u16)r;
}

union frag16 {
  uint4 q[2];
  v16bf v;
};

// Async 16B copy global -> LDS (GLOBAL_LOAD_ASYNC_TO_LDS_B128, ASYNCcnt)
__device__ __forceinline__ void async_copy_b128(const u16* gptr, u16* lptr) {
  unsigned lds = (unsigned)(size_t)lptr;  // low 32 bits = LDS offset
  unsigned long long ga = (unsigned long long)(size_t)gptr;
  asm volatile("global_load_async_to_lds_b128 %0, %1, off"
               :: "v"(lds), "v"(ga) : "memory");
}
template <int N>
__device__ __forceinline__ void wait_asynccnt() {
  asm volatile("s_wait_asynccnt %0" :: "i"(N) : "memory");
}
template <int N>
__device__ __forceinline__ void wait_tensorcnt() {
#if HAVE_TDM
  asm volatile("s_wait_tensorcnt %0" :: "i"(N) : "memory");
#endif
}

// TDM: load a 64(n) x 64(k) bf16 tile (row stride K elems) into LDS with
// hardware padding (32 DWORDs + 4 DWORDs) -> matches u16[64][72] layout.
__device__ __forceinline__ void tdm_load_tile(const u16* W, int K, int Ntot,
                                              int n0, int k0, unsigned lds_off) {
#if HAVE_TDM
  unsigned long long ga = (unsigned long long)(size_t)(W + (size_t)n0 * K + k0);
  u32x4 g0;
  g0[0] = 1u;                                              // count=1, user D#
  g0[1] = lds_off;                                         // lds_addr
  g0[2] = (u32)(ga & 0xFFFFFFFFull);                       // global_addr[31:0]
  g0[3] = (u32)((ga >> 32) & 0x1FFFFFFull) | (2u << 30);   // addr[56:32]|type=2
  i32x8 g1;
  g1[0] = (1 << 16)     // data_size = 2 bytes
        | (1 << 20)     // pad_enable
        | (4 << 22)     // pad_interval = 32 DWORDs (one 64-elem bf16 row)
        | (3 << 25);    // pad_amount   = 4 DWORDs (8 u16 pad)
  g1[1] = (K & 0xFFFF) << 16;                              // tensor_dim0 lo
  g1[2] = ((Ntot & 0xFFFF) << 16) | ((K >> 16) & 0xFFFF);  // dim1 lo | dim0 hi
  g1[3] = (64 << 16) | ((Ntot >> 16) & 0xFFFF);            // tile_dim0 | dim1 hi
  g1[4] = 64;                                              // tile_dim1 = 64 rows
  g1[5] = K;                                               // dim0_stride lo
  g1[6] = 0;
  g1[7] = 0;
  i32x4 z4 = {0, 0, 0, 0};
#if __clang_major__ >= 23
  i32x8 z8 = {0, 0, 0, 0, 0, 0, 0, 0};
  __builtin_amdgcn_tensor_load_to_lds(g0, g1, z4, z4, z8, 0);
#else
  __builtin_amdgcn_tensor_load_to_lds(g0, g1, z4, z4, 0);
#endif
#endif
}

// ---------------------------------------------------------------------------
// Transpose + convert input: (B,C,HW) f32 -> (B,HW,C) bf16   (read-coalesced)
// ---------------------------------------------------------------------------
__global__ __launch_bounds__(256) void transpose_in_kernel(
    const float* __restrict__ in, u16* __restrict__ out) {
  size_t i = (size_t)blockIdx.x * 256 + threadIdx.x;  // over B*C*HW
  if (i >= (size_t)BB * CCH * HWSZ) return;
  int hw = (int)(i & (HWSZ - 1));
  int c  = (int)((i >> 10) % CCH);
  int b  = (int)(i / ((size_t)CCH * HWSZ));
  out[((size_t)b * HWSZ + hw) * CCH + c] = f2bf(in[i]);
}

// ---------------------------------------------------------------------------
// Weight convert (keep native [N][K] layout): f32 -> bf16
// ---------------------------------------------------------------------------
__global__ __launch_bounds__(256) void wconv_kernel(
    const float* __restrict__ w, u16* __restrict__ wbf, int total) {
  int i = blockIdx.x * 256 + threadIdx.x;
  if (i >= total) return;
  wbf[i] = f2bf(w[i]);
}

// ---------------------------------------------------------------------------
// Tiled bf16 WMMA GEMM: C[M,N] = A[M,K] * W[N,K]^T + bias[N]
// Block tile 128x64, BK=64; 256 threads = 8 waves (4x2), each wave a 32x32
// patch -> 8 v_wmma per iteration (2 k-subblocks of 32). Double-buffered LDS;
// async-LDS copies (A) and TDM (B) for tile k+1 overlap WMMA on tile k.
// ---------------------------------------------------------------------------
__global__ __launch_bounds__(256) void gemm_bf16_kernel(
    const u16* __restrict__ A, const u16* __restrict__ W,
    const float* __restrict__ bias, float* __restrict__ Cmat,
    int M, int N, int K) {
  __shared__ __align__(16) u16 As[2][128][72];  // 2 x (128x64), 144B rows
  __shared__ __align__(16) u16 Bs[2][64][72];   // 2 x (64(n) x 64(k))

  const int tid  = threadIdx.x;
  const int lane = tid & 31;
  const int wid  = tid >> 5;
  const int half = lane >> 4;
  const int l16  = lane & 15;
  const int mgrp = wid >> 1;  // 0..3 -> 32-row group
  const int ngrp = wid & 1;   // 0..1 -> 32-col group

  // stage-in mapping: 32 rows x 8 threads x 16B, repeated over row blocks
  const int sr = tid >> 3, sc = (tid & 7) << 3;
  const size_t aBase = (size_t)(blockIdx.x * 128 + sr) * K;
  const size_t wBase = (size_t)(blockIdx.y * 64 + sr) * K;

  // fragment row/col indices (LDS)
  const int m0 = mgrp * 32 + l16, m1 = m0 + 16;
  const int n0 = ngrp * 32 + l16, n1 = n0 + 16;
  const int ka = half * 16;   // A chunk0 k-offset within 32-subblock
  const int kb = half * 16;   // B chunk0 k-offset within 32-subblock

  v8f acc00 = {}, acc01 = {}, acc10 = {}, acc11 = {};
  const int nIter = K / 64;

  auto stage = [&](int buf, int k0) {
#pragma unroll
    for (int rr = 0; rr < 4; ++rr)
      async_copy_b128(A + aBase + (size_t)(rr * 32) * K + k0 + sc,
                      &As[buf][sr + rr * 32][sc]);
#if HAVE_TDM
    if (wid == 0)
      tdm_load_tile(W, K, N, blockIdx.y * 64, k0,
                    (unsigned)(size_t)&Bs[buf][0][0]);
#else
#pragma unroll
    for (int rr = 0; rr < 2; ++rr)
      async_copy_b128(W + wBase + (size_t)(rr * 32) * K + k0 + sc,
                      &Bs[buf][sr + rr * 32][sc]);
#endif
  };

  stage(0, 0);  // prologue: tile 0 into buffer 0

  for (int it = 0; it < nIter; ++it) {
    const int p = it & 1;
    if (it + 1 < nIter) {
      // stage tile it+1 into the other buffer; overlaps this tile's WMMAs
      stage(p ^ 1, (it + 1) * 64);
      if (it + 2 < nIter)  // global_prefetch_b8 two tiles ahead
        __builtin_prefetch(A + aBase + (it + 1) * 64 + 64 + sc, 0, 3);
      wait_asynccnt<ASYNC_PER_ITER>();  // tile p landed (in-order counter)
      wait_tensorcnt<1>();
    } else {
      wait_asynccnt<0>();
      wait_tensorcnt<0>();
    }
    __syncthreads();  // barrier A: everyone's tile-p data visible

#pragma unroll
    for (int s = 0; s < 2; ++s) {  // two 32-K subblocks
      const int ks = s * 32;
      frag16 a0, a1, b0, b1;
      a0.q[0] = *(const uint4*)&As[p][m0][ks + ka];
      a0.q[1] = *(const uint4*)&As[p][m0][ks + ka + 8];
      a1.q[0] = *(const uint4*)&As[p][m1][ks + ka];
      a1.q[1] = *(const uint4*)&As[p][m1][ks + ka + 8];
      b0.q[0] = *(const uint4*)&Bs[p][n0][ks + kb];
      b0.q[1] = *(const uint4*)&Bs[p][n0][ks + kb + 8];
      b1.q[0] = *(const uint4*)&Bs[p][n1][ks + kb];
      b1.q[1] = *(const uint4*)&Bs[p][n1][ks + kb + 8];

      acc00 = __builtin_amdgcn_wmma_f32_16x16x32_bf16(
          false, a0.v, false, b0.v, (short)0, acc00, false, false);
      acc01 = __builtin_amdgcn_wmma_f32_16x16x32_bf16(
          false, a0.v, false, b1.v, (short)0, acc01, false, false);
      acc10 = __builtin_amdgcn_wmma_f32_16x16x32_bf16(
          false, a1.v, false, b0.v, (short)0, acc10, false, false);
      acc11 = __builtin_amdgcn_wmma_f32_16x16x32_bf16(
          false, a1.v, false, b1.v, (short)0, acc11, false, false);
    }

    __syncthreads();  // barrier B: all waves done reading tile p
  }

  // C/D layout: VGPR r -> M = r + 8*half ; N = lane&15
  const int gm0 = blockIdx.x * 128 + mgrp * 32 + half * 8;
  const int gm1 = gm0 + 16;
  const int gn0 = blockIdx.y * 64 + ngrp * 32 + l16;
  const int gn1 = gn0 + 16;
  const float bv0 = bias ? bias[gn0] : 0.f;
  const float bv1 = bias ? bias[gn1] : 0.f;
#pragma unroll
  for (int r = 0; r < 8; ++r) {
    Cmat[(size_t)(gm0 + r) * N + gn0] = acc00[r] + bv0;
    Cmat[(size_t)(gm0 + r) * N + gn1] = acc01[r] + bv1;
    Cmat[(size_t)(gm1 + r) * N + gn0] = acc10[r] + bv0;
    Cmat[(size_t)(gm1 + r) * N + gn1] = acc11[r] + bv1;
  }
}

// ---------------------------------------------------------------------------
// GroupNorm over conv1 output + concat-into-src + query(bf16) generation.
// ---------------------------------------------------------------------------
__global__ __launch_bounds__(256) void gn_src_kernel(
    const float* __restrict__ raw, const float* __restrict__ gamma,
    const float* __restrict__ beta, const float* __restrict__ level_embed,
    int lvl, float* __restrict__ src, u16* __restrict__ src_bf,
    u16* __restrict__ q_bf) {
  const int b = blockIdx.x >> 5;
  const int g = blockIdx.x & 31;
  const int tid = threadIdx.x;
  __shared__ float r1[256], r2[256];

  float s = 0.f, s2 = 0.f;
  for (int i = tid; i < 16 * HWSZ; i += 256) {
    int hw = i >> 4, dd = i & 15;
    float v = raw[((size_t)b * HWSZ + hw) * DMOD + g * 16 + dd];
    s += v; s2 += v * v;
  }
  r1[tid] = s; r2[tid] = s2;
  __syncthreads();
  for (int st = 128; st > 0; st >>= 1) {
    if (tid < st) { r1[tid] += r1[tid + st]; r2[tid] += r2[tid + st]; }
    __syncthreads();
  }
  const float mu = r1[0] * (1.f / 16384.f);
  const float var = r2[0] * (1.f / 16384.f) - mu * mu;
  const float rs = rsqrtf(var + 1e-5f);

  for (int i = tid; i < 16 * HWSZ; i += 256) {
    int hw = i >> 4, dd = i & 15;
    int d = g * 16 + dd;
    float v = raw[((size_t)b * HWSZ + hw) * DMOD + d];
    float n = (v - mu) * rs * gamma[d] + beta[d];
    size_t qi = ((size_t)b * LQ + (size_t)lvl * HWSZ + hw) * DMOD + d;
    src[qi] = n;
    src_bf[qi] = f2bf(n);
    q_bf[qi] = f2bf(n + level_embed[lvl * DMOD + d]);
  }
}

// ---------------------------------------------------------------------------
// Deformable-attention core: softmax(8) + bilinear sampling + weighted sum.
// ---------------------------------------------------------------------------
__global__ __launch_bounds__(256) void msda_kernel(
    const float* __restrict__ value, const float* __restrict__ offb,
    const float* __restrict__ awraw, u16* __restrict__ attn) {
  const int gw = blockIdx.x * 8 + (threadIdx.x >> 5);
  const int lane = threadIdx.x & 31;
  const int h = gw & 7;
  const int q = (gw >> 3) & (LQ - 1);
  const int b = gw >> 14;
  const int hw = q & (HWSZ - 1);
  const float bx = (float)(hw & 31);
  const float by = (float)(hw >> 5);

  const float* awp = awraw + ((size_t)b * LQ + q) * 64 + h * 8;
  float w[8], mx = -1e30f;
#pragma unroll
  for (int i = 0; i < 8; ++i) { w[i] = awp[i]; mx = fmaxf(mx, w[i]); }
  float sum = 0.f;
#pragma unroll
  for (int i = 0; i < 8; ++i) { w[i] = __expf(w[i] - mx); sum += w[i]; }
  const float inv = 1.f / sum;

  const float* op = offb + ((size_t)b * LQ + q) * 128 + h * 16;
  const int dh = lane * 2;
  float acc0 = 0.f, acc1 = 0.f;
#pragma unroll
  for (int l = 0; l < 2; ++l) {
    const float* vbase =
        value + (((size_t)b * 2 + l) * HWSZ) * DMOD + h * DHEAD + dh;
#pragma unroll
    for (int p = 0; p < 4; ++p) {
      const float x = bx + op[(l * 4 + p) * 2 + 0];
      const float y = by + op[(l * 4 + p) * 2 + 1];
      const float xf = floorf(x), yf = floorf(y);
      const float fx = x - xf, fy = y - yf;
      const int x0 = (int)xf, y0 = (int)yf;
      const float wt = w[l * 4 + p] * inv;
#pragma unroll
      for (int cy = 0; cy < 2; ++cy) {
#pragma unroll
        for (int cx = 0; cx < 2; ++cx) {
          const int xx = x0 + cx, yy = y0 + cy;
          if (xx >= 0 && xx < 32 && yy >= 0 && yy < 32) {
            const float cw = (cx ? fx : 1.f - fx) * (cy ? fy : 1.f - fy) * wt;
            const float* vp = vbase + (size_t)(yy * 32 + xx) * DMOD;
            acc0 += cw * vp[0];
            acc1 += cw * vp[1];
          }
        }
      }
    }
  }
  u16* o = attn + ((size_t)b * LQ + q) * DMOD + h * DHEAD + dh;
  o[0] = f2bf(acc0);
  o[1] = f2bf(acc1);
}

// ---------------------------------------------------------------------------
// Residual + LayerNorm over D=512 -> concat layout for final conv (bf16).
// ---------------------------------------------------------------------------
__global__ __launch_bounds__(256) void ln_kernel(
    const float* __restrict__ src, const float* __restrict__ proj,
    const float* __restrict__ g, const float* __restrict__ bta,
    u16* __restrict__ cat) {
  const int row = blockIdx.x;  // b*2048 + q
  const int b = row >> 11, q = row & (LQ - 1);
  const int tid = threadIdx.x;
  __shared__ float r1[256], r2[256];

  const size_t base = (size_t)row * DMOD;
  const float x0 = src[base + tid] + proj[base + tid];
  const float x1 = src[base + 256 + tid] + proj[base + 256 + tid];
  r1[tid] = x0 + x1;
  r2[tid] = x0 * x0 + x1 * x1;
  __syncthreads();
  for (int st = 128; st > 0; st >>= 1) {
    if (tid < st) { r1[tid] += r1[tid + st]; r2[tid] += r2[tid + st]; }
    __syncthreads();
  }
  const float mu = r1[0] * (1.f / 512.f);
  const float var = r2[0] * (1.f / 512.f) - mu * mu;
  const float rs = rsqrtf(var + 1e-5f);

  const int hw = q & (HWSZ - 1);
  const int coff = (q >> 10) * DMOD;
  const size_t ob = ((size_t)b * HWSZ + hw) * (2 * DMOD) + coff;
  cat[ob + tid]       = f2bf((x0 - mu) * rs * g[tid] + bta[tid]);
  cat[ob + 256 + tid] = f2bf((x1 - mu) * rs * g[256 + tid] + bta[256 + tid]);
}

// ---------------------------------------------------------------------------
// Final GroupNorm: conv:[B][HW][768] f32 -> out:[B][768][HW] f32 (NCHW)
// ---------------------------------------------------------------------------
__global__ __launch_bounds__(256) void gn_final_kernel(
    const float* __restrict__ conv, const float* __restrict__ gc,
    const float* __restrict__ bec, float* __restrict__ out) {
  const int b = blockIdx.x >> 5;
  const int g = blockIdx.x & 31;
  const int tid = threadIdx.x;
  __shared__ float r1[256], r2[256];

  float s = 0.f, s2 = 0.f;
  for (int i = tid; i < 24 * HWSZ; i += 256) {
    int hw = i / 24, cc = i % 24;
    float v = conv[((size_t)b * HWSZ + hw) * CCH + g * 24 + cc];
    s += v; s2 += v * v;
  }
  r1[tid] = s; r2[tid] = s2;
  __syncthreads();
  for (int st = 128; st > 0; st >>= 1) {
    if (tid < st) { r1[tid] += r1[tid + st]; r2[tid] += r2[tid + st]; }
    __syncthreads();
  }
  const float mu = r1[0] * (1.f / 24576.f);
  const float var = r2[0] * (1.f / 24576.f) - mu * mu;
  const float rs = rsqrtf(var + 1e-5f);

  for (int i = tid; i < 24 * HWSZ; i += 256) {
    int hw = i / 24, cc = i % 24;
    int c = g * 24 + cc;
    float v = conv[((size_t)b * HWSZ + hw) * CCH + c];
    out[((size_t)b * CCH + c) * HWSZ + hw] = (v - mu) * rs * gc[c] + bec[c];
  }
}

// ---------------------------------------------------------------------------
extern "C" void kernel_launch(void* const* d_in, const int* in_sizes, int n_in,
                              void* d_out, int out_size, void* d_ws,
                              size_t ws_size, hipStream_t stream) {
  const float* input_v     = (const float*)d_in[0];
  const float* input_i     = (const float*)d_in[1];
  const float* wv          = (const float*)d_in[2];
  const float* bv          = (const float*)d_in[3];
  const float* gv          = (const float*)d_in[4];
  const float* bev         = (const float*)d_in[5];
  const float* wi          = (const float*)d_in[6];
  const float* bi          = (const float*)d_in[7];
  const float* gi          = (const float*)d_in[8];
  const float* bei         = (const float*)d_in[9];
  const float* level_embed = (const float*)d_in[10];
  const float* w_off       = (const float*)d_in[11];
  const float* b_off       = (const float*)d_in[12];
  const float* w_attn      = (const float*)d_in[13];
  const float* b_attn      = (const float*)d_in[14];
  const float* w_val       = (const float*)d_in[15];
  const float* b_val       = (const float*)d_in[16];
  const float* w_out       = (const float*)d_in[17];
  const float* b_out       = (const float*)d_in[18];
  const float* ln_g        = (const float*)d_in[19];
  const float* ln_b        = (const float*)d_in[20];
  const float* wc          = (const float*)d_in[21];
  const float* bc          = (const float*)d_in[22];
  const float* gc          = (const float*)d_in[23];
  const float* bec         = (const float*)d_in[24];
  float* out = (float*)d_out;

  char* ws = (char*)d_ws;
  size_t off = 0;
  auto alloc = [&](size_t bytes) -> size_t {
    size_t r = off;
    off += (bytes + 255) & ~(size_t)255;
    return r;
  };
  const size_t o_inv   = alloc((size_t)BB * HWSZ * CCH * 2);
  const size_t o_ini   = alloc((size_t)BB * HWSZ * CCH * 2);
  const size_t o_wv    = alloc((size_t)DMOD * CCH * 2);
  const size_t o_wi    = alloc((size_t)DMOD * CCH * 2);
  const size_t o_wval  = alloc((size_t)DMOD * DMOD * 2);
  const size_t o_woff  = alloc((size_t)128 * DMOD * 2);
  const size_t o_wat   = alloc((size_t)64 * DMOD * 2);
  const size_t o_wout  = alloc((size_t)DMOD * DMOD * 2);
  const size_t o_wc    = alloc((size_t)CCH * 2 * DMOD * 2);
  const size_t o_rawv  = alloc((size_t)BB * HWSZ * DMOD * 4);
  const size_t o_rawi  = alloc((size_t)BB * HWSZ * DMOD * 4);
  const size_t o_src   = alloc((size_t)BB * LQ * DMOD * 4);
  const size_t o_srcbf = alloc((size_t)BB * LQ * DMOD * 2);
  const size_t o_qbf   = alloc((size_t)BB * LQ * DMOD * 2);
  const size_t o_value = alloc((size_t)BB * LQ * DMOD * 4);
  const size_t o_proj  = alloc((size_t)BB * LQ * DMOD * 4);
  // aliases (producers dead at time of reuse):
  const size_t o_offb  = o_rawv;  // [B][2048][128] f32
  const size_t o_awraw = o_rawi;  // [B][2048][64]  f32
  const size_t o_attn  = o_inv;   // [B][2048][512] bf16
  const size_t o_cat   = o_qbf;   // [B][1024][1024] bf16
  const size_t o_conv2 = o_src;   // [B][1024][768] f32

  u16*   inv_bf  = (u16*)(ws + o_inv);
  u16*   ini_bf  = (u16*)(ws + o_ini);
  u16*   wv_bf   = (u16*)(ws + o_wv);
  u16*   wi_bf   = (u16*)(ws + o_wi);
  u16*   wval_bf = (u16*)(ws + o_wval);
  u16*   woff_bf = (u16*)(ws + o_woff);
  u16*   wat_bf  = (u16*)(ws + o_wat);
  u16*   wout_bf = (u16*)(ws + o_wout);
  u16*   wc_bf   = (u16*)(ws + o_wc);
  float* rawv    = (float*)(ws + o_rawv);
  float* rawi    = (float*)(ws + o_rawi);
  float* src     = (float*)(ws + o_src);
  u16*   src_bf  = (u16*)(ws + o_srcbf);
  u16*   q_bf    = (u16*)(ws + o_qbf);
  float* value   = (float*)(ws + o_value);
  float* projbuf = (float*)(ws + o_proj);
  float* offb    = (float*)(ws + o_offb);
  float* awraw   = (float*)(ws + o_awraw);
  u16*   attn_bf = (u16*)(ws + o_attn);
  u16*   cat_bf  = (u16*)(ws + o_cat);
  float* conv2   = (float*)(ws + o_conv2);

  const dim3 blk(256);

  // 1) input transpose/convert
  transpose_in_kernel<<<24576, blk, 0, stream>>>(input_v, inv_bf);
  transpose_in_kernel<<<24576, blk, 0, stream>>>(input_i, ini_bf);

  // 2) weight convert (native [N][K] layout)
  wconv_kernel<<<(DMOD * CCH) / 256, blk, 0, stream>>>(wv, wv_bf, DMOD * CCH);
  wconv_kernel<<<(DMOD * CCH) / 256, blk, 0, stream>>>(wi, wi_bf, DMOD * CCH);
  wconv_kernel<<<(DMOD * DMOD) / 256, blk, 0, stream>>>(w_val, wval_bf, DMOD * DMOD);
  wconv_kernel<<<(128 * DMOD) / 256, blk, 0, stream>>>(w_off, woff_bf, 128 * DMOD);
  wconv_kernel<<<(64 * DMOD) / 256, blk, 0, stream>>>(w_attn, wat_bf, 64 * DMOD);
  wconv_kernel<<<(DMOD * DMOD) / 256, blk, 0, stream>>>(w_out, wout_bf, DMOD * DMOD);
  wconv_kernel<<<(CCH * 2 * DMOD) / 256, blk, 0, stream>>>(wc, wc_bf, CCH * 2 * DMOD);

  // 3) conv1x1 GEMMs (8192 x 512 x 768)
  gemm_bf16_kernel<<<dim3(8192 / 128, DMOD / 64), blk, 0, stream>>>(
      inv_bf, wv_bf, bv, rawv, 8192, DMOD, CCH);
  gemm_bf16_kernel<<<dim3(8192 / 128, DMOD / 64), blk, 0, stream>>>(
      ini_bf, wi_bf, bi, rawi, 8192, DMOD, CCH);

  // 4) GroupNorm + src/query build
  gn_src_kernel<<<BB * 32, blk, 0, stream>>>(rawv, gv, bev, level_embed, 0,
                                             src, src_bf, q_bf);
  gn_src_kernel<<<BB * 32, blk, 0, stream>>>(rawi, gi, bei, level_embed, 1,
                                             src, src_bf, q_bf);

  // 5) attention projections (16384 rows)
  gemm_bf16_kernel<<<dim3(16384 / 128, DMOD / 64), blk, 0, stream>>>(
      src_bf, wval_bf, b_val, value, 16384, DMOD, DMOD);
  gemm_bf16_kernel<<<dim3(16384 / 128, 128 / 64), blk, 0, stream>>>(
      q_bf, woff_bf, b_off, offb, 16384, 128, DMOD);
  gemm_bf16_kernel<<<dim3(16384 / 128, 64 / 64), blk, 0, stream>>>(
      q_bf, wat_bf, b_attn, awraw, 16384, 64, DMOD);

  // 6) softmax + bilinear sampling + weighted sum
  msda_kernel<<<16384, blk, 0, stream>>>(value, offb, awraw, attn_bf);

  // 7) output projection
  gemm_bf16_kernel<<<dim3(16384 / 128, DMOD / 64), blk, 0, stream>>>(
      attn_bf, wout_bf, b_out, projbuf, 16384, DMOD, DMOD);

  // 8) residual + LayerNorm -> concat layout
  ln_kernel<<<BB * LQ, blk, 0, stream>>>(src, projbuf, ln_g, ln_b, cat_bf);

  // 9) final conv1x1 (8192 x 768 x 1024)
  gemm_bf16_kernel<<<dim3(8192 / 128, CCH / 64), blk, 0, stream>>>(
      cat_bf, wc_bf, bc, conv2, 8192, CCH, 2 * DMOD);

  // 10) final GroupNorm -> NCHW output
  gn_final_kernel<<<BB * 32, blk, 0, stream>>>(conv2, gc, bec, out);
}
